// EncoderBlock_64123861729329
// MI455X (gfx1250) — matrix operand order repaired and checked by reading, not compiled, not run on hardware
//
#include <hip/hip_runtime.h>
#include <hip/hip_bf16.h>
#include <math.h>

// ---------------------------------------------------------------------------
// Types
// ---------------------------------------------------------------------------
typedef _Float16 half_t;
typedef __attribute__((ext_vector_type(16))) _Float16 v16h;
typedef __attribute__((ext_vector_type(8)))  _Float16 v8h;
typedef __attribute__((ext_vector_type(8)))  float    v8f;

#define WMMA_F16(a, b, c) \
    __builtin_amdgcn_wmma_f32_16x16x32_f16(false, (a), false, (b), (short)0, (c), false, false)

// Tensor Data Mover availability (this toolchain: clang-23 6-arg builtin)
#if defined(__gfx1250__) && __has_builtin(__builtin_amdgcn_tensor_load_to_lds) && \
    __has_builtin(__builtin_amdgcn_s_wait_tensorcnt)
#define HAVE_TDM 1
#else
#define HAVE_TDM 0
#endif

// Problem constants
static constexpr int Bdim = 4;
static constexpr int Sdim = 2048;
static constexpr int Ddim = 512;
static constexpr int Hdim = 8;
static constexpr int HDdim = 64;
static constexpr int DFdim = 2048;
static constexpr int ROWS = Bdim * Sdim;      // 8192

__device__ __forceinline__ v16h mk_frag(v8h lo, v8h hi) {
    v16h r;
#pragma unroll
    for (int i = 0; i < 8; ++i) { r[i] = lo[i]; r[i + 8] = hi[i]; }
    return r;
}

#if HAVE_TDM
typedef __attribute__((ext_vector_type(4))) unsigned int u32x4;
typedef __attribute__((ext_vector_type(8))) int          i32x8;
typedef __attribute__((ext_vector_type(4))) int          i32x4;

// Issue a 2D TDM load: tile_h rows x tile_w f16 elements, row stride
// row_stride_elems, packed row-major into LDS at lds_byte_off.
// D# layout per CDNA5 ISA 8.3/8.4 (group0 128b, group1 256b; groups 2/3 zero
// => 2D tensor).
__device__ __forceinline__ void tdm_load_2d(const half_t* gsrc, unsigned lds_byte_off,
                                            unsigned tile_w, unsigned tile_h,
                                            unsigned tens_w, unsigned tens_h,
                                            unsigned row_stride_elems) {
    unsigned long long ga = (unsigned long long)(size_t)gsrc;
    u32x4 g0;
    g0[0] = 1u;                                             // count=1 user descriptor
    g0[1] = lds_byte_off;                                   // lds_addr
    g0[2] = (unsigned)(ga & 0xFFFFFFFFu);                   // global_addr[31:0]
    g0[3] = (unsigned)((ga >> 32) & 0x01FFFFFFu)            // global_addr[56:32]
          | (2u << 30);                                     // type = 2 ("image")
    i32x8 g1;
    g1[0] = (int)(1u << 16);                                // data_size=1 -> 2 bytes
    g1[1] = (int)((tens_w & 0xFFFFu) << 16);                // tensor_dim0[15:0]
    g1[2] = (int)(((tens_w >> 16) & 0xFFFFu)                // tensor_dim0[31:16]
                | ((tens_h & 0xFFFFu) << 16));              // tensor_dim1[15:0]
    g1[3] = (int)(((tens_h >> 16) & 0xFFFFu)                // tensor_dim1[31:16]
                | ((tile_w & 0xFFFFu) << 16));              // tile_dim0
    g1[4] = (int)(tile_h & 0xFFFFu);                        // tile_dim1 (tile_dim2=0)
    g1[5] = (int)row_stride_elems;                          // tensor_dim0_stride[31:0]
    g1[6] = 0;                                              // stride[47:32] | dim1_stride lo
    g1[7] = 0;
    i32x4 z4 = {0, 0, 0, 0};
    i32x8 z8 = {0, 0, 0, 0, 0, 0, 0, 0};
    __builtin_amdgcn_tensor_load_to_lds(g0, g1, z4, z4, z8, 0);
}
#endif

// ---------------------------------------------------------------------------
// f32 -> f16 conversion
// ---------------------------------------------------------------------------
__global__ void cvt_f32_to_f16(const float* __restrict__ src,
                               half_t* __restrict__ dst, int n) {
    int i = blockIdx.x * blockDim.x + threadIdx.x;
    if (i < n) dst[i] = (half_t)src[i];
}

// ---------------------------------------------------------------------------
// Tiled GEMM:  C[M,N] = A[M,K] (f16, row major) * W[K,N] (f16, row major) + bias
// Block tile 128x128, K step 32, 256 threads = 8 waves (2x4 wave grid),
// each wave computes 64x32 via 4x2 WMMA fragments.
// A tile staged by the Tensor Data Mover when available.
// EPI: 0 = f32 out, 1 = f16 out with ReLU, 2 = f16 out
// ---------------------------------------------------------------------------
template <int EPI>
__global__ __launch_bounds__(256)
void gemm_kernel(const half_t* __restrict__ A, const half_t* __restrict__ W,
                 const float* __restrict__ bias,
                 float* __restrict__ Cf, half_t* __restrict__ Ch,
                 int M, int N, int K) {
    __shared__ __align__(16) half_t As[128 * 32];   // [m][k]
    __shared__ __align__(16) half_t Bs[128 * 32];   // transposed: [n][k]

    const int tid  = threadIdx.x;
    const int wave = tid >> 5;
    const int lane = tid & 31;
    const int lh   = lane & 15;
    const bool hiL = lane >= 16;
    const int wm   = wave >> 2;       // 0..1
    const int wn   = wave & 3;        // 0..3
    const int bm   = blockIdx.y * 128;
    const int bn   = blockIdx.x * 128;

    v8f acc[4][2] = {};

    // B tile load indices: 8 threads per k-row, 16 halves each
    const int bk = tid >> 3, bn0 = (tid & 7) * 16;

    for (int k0 = 0; k0 < K; k0 += 32) {
        // ---- stage A tile ----
#if HAVE_TDM
        if (wave == 0) {
            tdm_load_2d(A + (size_t)bm * K + k0, (unsigned)(size_t)(void*)As,
                        /*tile_w=*/32, /*tile_h=*/128,
                        /*tens_w=*/(unsigned)(K - k0), /*tens_h=*/(unsigned)(M - bm),
                        /*stride=*/(unsigned)K);
        }
#else
        {
            const int arow = tid >> 1, aoff = (tid & 1) * 16;
            const half_t* src = A + (size_t)(bm + arow) * K + k0 + aoff;
            *(v8h*)&As[arow * 32 + aoff]     = *(const v8h*)src;
            *(v8h*)&As[arow * 32 + aoff + 8] = *(const v8h*)(src + 8);
            if (k0 + 32 < K) __builtin_prefetch(src + 32, 0, 1);
        }
#endif
        // ---- stage B tile (transposed into LDS) ----
        {
            const half_t* src = W + (size_t)(k0 + bk) * N + bn + bn0;
            v8h t0 = *(const v8h*)src;
            v8h t1 = *(const v8h*)(src + 8);
            if (k0 + 32 < K) __builtin_prefetch(src + (size_t)32 * N, 0, 1);
#pragma unroll
            for (int i = 0; i < 8; ++i) {
                Bs[(bn0 + i) * 32 + bk]     = t0[i];
                Bs[(bn0 + 8 + i) * 32 + bk] = t1[i];
            }
        }
#if HAVE_TDM
        if (wave == 0) __builtin_amdgcn_s_wait_tensorcnt(0);
#endif
        __syncthreads();

        // ---- build A fragments (16x32 each) ----
        v16h af[4];
        {
            const int kb = hiL ? 8 : 0;
#pragma unroll
            for (int fm = 0; fm < 4; ++fm) {
                const int row = wm * 64 + fm * 16 + lh;
                v8h lo = *(const v8h*)&As[row * 32 + kb];
                v8h hi = *(const v8h*)&As[row * 32 + kb + 16];
                af[fm] = mk_frag(lo, hi);
            }
        }
        // ---- build B fragments (32x16 each) ----
        v16h bf[2];
        {
            const int kb = hiL ? 8 : 0;
#pragma unroll
            for (int fn = 0; fn < 2; ++fn) {
                const int col = wn * 32 + fn * 16 + lh;
                v8h lo = *(const v8h*)&Bs[col * 32 + kb];
                v8h hi = *(const v8h*)&Bs[col * 32 + kb + 16];
                bf[fn] = mk_frag(lo, hi);
            }
        }
#pragma unroll
        for (int fm = 0; fm < 4; ++fm)
#pragma unroll
            for (int fn = 0; fn < 2; ++fn)
                acc[fm][fn] = WMMA_F16(af[fm], bf[fn], acc[fm][fn]);
        __syncthreads();
    }

    // ---- epilogue ----
    const int rbase = hiL ? 8 : 0;
#pragma unroll
    for (int fn = 0; fn < 2; ++fn) {
        const int col = bn + wn * 32 + fn * 16 + lh;
        const float bv = bias ? bias[col] : 0.0f;
#pragma unroll
        for (int fm = 0; fm < 4; ++fm) {
#pragma unroll
            for (int r = 0; r < 8; ++r) {
                const int row = bm + wm * 64 + fm * 16 + rbase + r;
                float v = acc[fm][fn][r] + bv;
                if (EPI == 0) {
                    Cf[(size_t)row * N + col] = v;
                } else if (EPI == 1) {
                    Ch[(size_t)row * N + col] = (half_t)fmaxf(v, 0.0f);
                } else {
                    Ch[(size_t)row * N + col] = (half_t)v;
                }
            }
        }
    }
}

// ---------------------------------------------------------------------------
// Flash attention with ALiBi.  One block = (b, h, 64 q rows).  128 threads.
// Q,K,V: f16 [B*S, D] with head h at columns [h*64, h*64+64).
// ---------------------------------------------------------------------------
__global__ __launch_bounds__(128)
void attn_kernel(const half_t* __restrict__ Q, const half_t* __restrict__ Km,
                 const half_t* __restrict__ Vm, const int* __restrict__ mask,
                 half_t* __restrict__ AO) {
    __shared__ __align__(16) half_t Qs[64 * 64];
    __shared__ __align__(16) half_t Ks[64 * 64];   // [k][d]
    __shared__ __align__(16) half_t Vs[64 * 64];   // transposed: [d][k]
    __shared__ __align__(16) float  Sf[64 * 64];
    __shared__ __align__(16) half_t Ps[64 * 64];
    __shared__ float mrow[64], lrow[64], frow[64];

    const int tid  = threadIdx.x;
    const int wave = tid >> 5;
    const int lane = tid & 31;
    const int lh   = lane & 15;
    const bool hiL = lane >= 16;
    const int rbase = hiL ? 8 : 0;

    const int b  = blockIdx.z;
    const int h  = blockIdx.y;
    const int q0 = blockIdx.x * 64;

    const float scale = 0.125f;                       // HD^-0.5
    const float slope = exp2f(-((float)h) / (float)Hdim);
    const float NEG_INF = -__builtin_inff();

    const size_t headoff = (size_t)h * HDdim;
    const half_t* Qb = Q + (size_t)b * Sdim * Ddim + headoff;
    const half_t* Kb = Km + (size_t)b * Sdim * Ddim + headoff;
    const half_t* Vb = Vm + (size_t)b * Sdim * Ddim + headoff;
    const int* mb = mask + (size_t)b * Sdim;

    // ---- load Q tile (64 x 64) ----
    {
        const int row = tid >> 1, off = (tid & 1) * 32;
        const half_t* src = Qb + (size_t)(q0 + row) * Ddim + off;
#pragma unroll
        for (int i = 0; i < 4; ++i)
            *(v8h*)&Qs[row * 64 + off + i * 8] = *(const v8h*)(src + i * 8);
    }
    if (tid < 64) { mrow[tid] = NEG_INF; lrow[tid] = 0.0f; }
    __syncthreads();

    // ---- Q fragments: 2 per wave (d = 0..31, 32..63) ----
    v16h qa[2];
#pragma unroll
    for (int kd = 0; kd < 2; ++kd) {
        const int row = wave * 16 + lh;
        const int kb = kd * 32 + (hiL ? 8 : 0);
        v8h lo = *(const v8h*)&Qs[row * 64 + kb];
        v8h hi = *(const v8h*)&Qs[row * 64 + kb + 16];
        qa[kd] = mk_frag(lo, hi);
    }

    v8f oacc[4] = {};

    for (int kt = 0; kt < Sdim / 64; ++kt) {
        const int kbase = kt * 64;
        // ---- stage K tile (row major) and V tile (transposed) ----
        {
            const int row = tid >> 1, off = (tid & 1) * 32;
            const half_t* ks = Kb + (size_t)(kbase + row) * Ddim + off;
#pragma unroll
            for (int i = 0; i < 4; ++i)
                *(v8h*)&Ks[row * 64 + off + i * 8] = *(const v8h*)(ks + i * 8);
            const half_t* vs = Vb + (size_t)(kbase + row) * Ddim + off;
#pragma unroll
            for (int i = 0; i < 4; ++i) {
                v8h t = *(const v8h*)(vs + i * 8);
#pragma unroll
                for (int j = 0; j < 8; ++j)
                    Vs[(off + i * 8 + j) * 64 + row] = t[j];
            }
        }
        __syncthreads();

        // ---- scores: 4 fragments of 16x16 per wave ----
#pragma unroll
        for (int fn = 0; fn < 4; ++fn) {
            v8f s = {};
#pragma unroll
            for (int kd = 0; kd < 2; ++kd) {
                const int col = fn * 16 + lh;
                const int kb = kd * 32 + (hiL ? 8 : 0);
                v8h lo = *(const v8h*)&Ks[col * 64 + kb];
                v8h hi = *(const v8h*)&Ks[col * 64 + kb + 16];
                s = WMMA_F16(qa[kd], mk_frag(lo, hi), s);
            }
#pragma unroll
            for (int r = 0; r < 8; ++r)
                Sf[(wave * 16 + rbase + r) * 64 + fn * 16 + lh] = s[r];
        }
        __syncthreads();

        // ---- online softmax (one thread per q row) ----
        if (tid < 64) {
            const int qg = q0 + tid;
            const float mo = mrow[tid];
            float mx = mo;
            for (int j = 0; j < 64; ++j) {
                float s = Sf[tid * 64 + j] * scale
                        + slope * (float)((kbase + j) - qg);
                if (mb[kbase + j] == 0) s = NEG_INF;
                Sf[tid * 64 + j] = s;
                mx = fmaxf(mx, s);
            }
            const float f = (mo == NEG_INF) ? 0.0f : __expf(mo - mx);
            float sum = 0.0f;
            for (int j = 0; j < 64; ++j) {
                const float s = Sf[tid * 64 + j];
                const float p = (s == NEG_INF) ? 0.0f : __expf(s - mx);
                Ps[tid * 64 + j] = (half_t)p;
                sum += p;
            }
            frow[tid] = f;
            mrow[tid] = mx;
            lrow[tid] = lrow[tid] * f + sum;
        }
        __syncthreads();

        // ---- rescale output accumulators ----
        float fr[8];
#pragma unroll
        for (int r = 0; r < 8; ++r) fr[r] = frow[wave * 16 + rbase + r];
#pragma unroll
        for (int fn = 0; fn < 4; ++fn)
#pragma unroll
            for (int r = 0; r < 8; ++r) oacc[fn][r] *= fr[r];

        // ---- P fragments (16x32 each) ----
        v16h pa[2];
#pragma unroll
        for (int kd = 0; kd < 2; ++kd) {
            const int row = wave * 16 + lh;
            const int kb = kd * 32 + (hiL ? 8 : 0);
            v8h lo = *(const v8h*)&Ps[row * 64 + kb];
            v8h hi = *(const v8h*)&Ps[row * 64 + kb + 16];
            pa[kd] = mk_frag(lo, hi);
        }
        // ---- O += P * V ----
#pragma unroll
        for (int fn = 0; fn < 4; ++fn) {
#pragma unroll
            for (int kd = 0; kd < 2; ++kd) {
                const int col = fn * 16 + lh;
                const int kb = kd * 32 + (hiL ? 8 : 0);
                v8h lo = *(const v8h*)&Vs[col * 64 + kb];
                v8h hi = *(const v8h*)&Vs[col * 64 + kb + 16];
                oacc[fn] = WMMA_F16(pa[kd], mk_frag(lo, hi), oacc[fn]);
            }
        }
        __syncthreads();
    }

    // ---- finalize: O / l, store f16 ----
#pragma unroll
    for (int fn = 0; fn < 4; ++fn) {
        const int col = h * HDdim + fn * 16 + lh;
#pragma unroll
        for (int r = 0; r < 8; ++r) {
            const int rr = wave * 16 + rbase + r;
            const float l = lrow[rr];
            const float v = (l > 0.0f) ? oacc[fn][r] / l : 0.0f;
            AO[((size_t)b * Sdim + q0 + rr) * Ddim + col] = (half_t)v;
        }
    }
}

// ---------------------------------------------------------------------------
// out = LayerNorm(X + Y) * g + b ; optional f16 copy.  One block per row.
// ---------------------------------------------------------------------------
__global__ __launch_bounds__(256)
void add_ln_kernel(const float* __restrict__ X, const float* __restrict__ Y,
                   const float* __restrict__ g, const float* __restrict__ bb,
                   float* __restrict__ out, half_t* __restrict__ outh) {
    __shared__ float rs[256], rq[256];
    const int row = blockIdx.x, tid = threadIdx.x;
    const size_t base = (size_t)row * Ddim;

    const float v0 = X[base + tid] + Y[base + tid];
    const float v1 = X[base + tid + 256] + Y[base + tid + 256];
    rs[tid] = v0 + v1;
    rq[tid] = v0 * v0 + v1 * v1;
    __syncthreads();
    for (int s = 128; s > 0; s >>= 1) {
        if (tid < s) { rs[tid] += rs[tid + s]; rq[tid] += rq[tid + s]; }
        __syncthreads();
    }
    const float mu   = rs[0] * (1.0f / Ddim);
    const float var  = rq[0] * (1.0f / Ddim) - mu * mu;
    const float rinv = rsqrtf(var + 1e-5f);
    const float o0 = g[tid] * (v0 - mu) * rinv + bb[tid];
    const float o1 = g[tid + 256] * (v1 - mu) * rinv + bb[tid + 256];
    out[base + tid]       = o0;
    out[base + tid + 256] = o1;
    if (outh) {
        outh[base + tid]       = (half_t)o0;
        outh[base + tid + 256] = (half_t)o1;
    }
}

// ---------------------------------------------------------------------------
// Orchestration
// ---------------------------------------------------------------------------
extern "C" void kernel_launch(void* const* d_in, const int* in_sizes, int n_in,
                              void* d_out, int out_size, void* d_ws, size_t ws_size,
                              hipStream_t stream) {
    const float* x    = (const float*)d_in[0];
    const int*   mask = (const int*)d_in[1];
    const float* Wq = (const float*)d_in[2];  const float* bq = (const float*)d_in[3];
    const float* Wk = (const float*)d_in[4];  const float* bk = (const float*)d_in[5];
    const float* Wv = (const float*)d_in[6];  const float* bv = (const float*)d_in[7];
    const float* Wo = (const float*)d_in[8];  const float* bo = (const float*)d_in[9];
    const float* Wh = (const float*)d_in[10]; const float* bh = (const float*)d_in[11];
    const float* Wf = (const float*)d_in[12]; const float* bf = (const float*)d_in[13];
    const float* g1 = (const float*)d_in[14]; const float* b1 = (const float*)d_in[15];
    const float* g2 = (const float*)d_in[16]; const float* b2 = (const float*)d_in[17];
    float* out = (float*)d_out;

    char* ws = (char*)d_ws;
    // workspace layout (bytes)
    half_t* xh  = (half_t*)(ws + 0);                       //  8 MB
    half_t* Wqh = (half_t*)(ws + 8388608);                 //  512 KB
    half_t* Wkh = (half_t*)(ws + 8912896);
    half_t* Wvh = (half_t*)(ws + 9437184);
    half_t* Woh = (half_t*)(ws + 9961472);
    half_t* Whh = (half_t*)(ws + 10485760);                //  2 MB
    half_t* Wfh = (half_t*)(ws + 12582912);                //  2 MB
    half_t* Qh  = (half_t*)(ws + 14680064);                //  8 MB
    half_t* Kh  = (half_t*)(ws + 23068672);                //  8 MB
    half_t* Vh  = (half_t*)(ws + 31457280);                //  8 MB
    half_t* AOh = (half_t*)(ws + 39845888);                //  8 MB
    half_t* Hh  = (half_t*)(ws + 14680064);                // 32 MB, overlays Q/K/V/AO
    float*  tmp = (float*)(ws + 48234496);                 // 16 MB
    float*  x1  = (float*)(ws + 65011712);                 // 16 MB
    half_t* x1h = (half_t*)(ws + 81788928);                //  8 MB

    // 1) conversions to f16
    auto cvt = [&](const float* s, half_t* d, int n) {
        cvt_f32_to_f16<<<(n + 255) / 256, 256, 0, stream>>>(s, d, n);
    };
    cvt(x,  xh,  ROWS * Ddim);
    cvt(Wq, Wqh, Ddim * Ddim);
    cvt(Wk, Wkh, Ddim * Ddim);
    cvt(Wv, Wvh, Ddim * Ddim);
    cvt(Wo, Woh, Ddim * Ddim);
    cvt(Wh, Whh, Ddim * DFdim);
    cvt(Wf, Wfh, DFdim * Ddim);

    // 2) Q / K / V projections -> f16
    dim3 blk(256);
    dim3 gQKV(Ddim / 128, ROWS / 128);
    gemm_kernel<2><<<gQKV, blk, 0, stream>>>(xh, Wqh, bq, nullptr, Qh, ROWS, Ddim, Ddim);
    gemm_kernel<2><<<gQKV, blk, 0, stream>>>(xh, Wkh, bk, nullptr, Kh, ROWS, Ddim, Ddim);
    gemm_kernel<2><<<gQKV, blk, 0, stream>>>(xh, Wvh, bv, nullptr, Vh, ROWS, Ddim, Ddim);

    // 3) flash attention with ALiBi
    attn_kernel<<<dim3(Sdim / 64, Hdim, Bdim), dim3(128), 0, stream>>>(
        Qh, Kh, Vh, mask, AOh);

    // 4) output projection -> f32
    gemm_kernel<0><<<gQKV, blk, 0, stream>>>(AOh, Woh, bo, tmp, nullptr, ROWS, Ddim, Ddim);

    // 5) x1 = LN(x + attn_out)
    add_ln_kernel<<<ROWS, 256, 0, stream>>>(x, tmp, g1, b1, x1, x1h);

    // 6) FFN
    dim3 gF1(DFdim / 128, ROWS / 128);
    gemm_kernel<1><<<gF1, blk, 0, stream>>>(x1h, Whh, bh, nullptr, Hh, ROWS, DFdim, Ddim);
    gemm_kernel<0><<<gQKV, blk, 0, stream>>>(Hh, Wfh, bf, tmp, nullptr, ROWS, Ddim, DFdim);

    // 7) out = LN(x1 + ffn)
    add_ln_kernel<<<ROWS, 256, 0, stream>>>(x1, tmp, g2, b2, out, nullptr);
}
